// cola_gnn_31507880084152
// MI455X (gfx1250) — matrix operand order repaired and checked by reading, not compiled, not run on hardware
//
#include <hip/hip_runtime.h>
#include <hip/hip_bf16.h>

// ---------------------------------------------------------------------------
// cola-gnn pipeline for MI455X (gfx1250, wave32, WMMA).
// All dense GEMMs run on v_wmma_f32_16x16x32_f16 (f16 in, f32 accumulate).
// ---------------------------------------------------------------------------

typedef _Float16 f16;
typedef __attribute__((ext_vector_type(16))) _Float16 v16h;
typedef __attribute__((ext_vector_type(8)))  float    v8f;

#define Bc     16
#define Wc     32
#define Mc     512
#define NHc    64
#define HALFc  32
#define Kc     8
#define Hc     8
#define ROWSc  (Bc * Mc)          // 8192

// ---- fast scalar helpers (branch-free; keep EXEC uniform around WMMA) ------

__device__ __forceinline__ float rcp_fast(float x) {
  return __builtin_amdgcn_rcpf(x);
}

// tanh(x) = (e^{2x}-1)/(e^{2x}+1); clamp keeps exp finite -> no NaN, no branch
__device__ __forceinline__ float tanh_fast(float x) {
  float xc = fminf(fmaxf(x, -10.f), 10.f);
  float e  = __expf(2.f * xc);
  return (e - 1.f) * rcp_fast(e + 1.f);
}

__device__ __forceinline__ float sigmoid_fast(float x) {
  return rcp_fast(1.f + __expf(-x));
}

// ---- WMMA fragment helpers (layouts per CDNA5 ISA 7.12.2, wave32) ----------

// A fragment: 16x32 f16, row-major source with leading dim lda.
// lane l (0..15): row l, K = {0..7, 16..23}; lane l+16: row l, K = {8..15, 24..31}
__device__ __forceinline__ v16h frag_load_A(const f16* __restrict__ s, int lda) {
  const int l    = threadIdx.x & 31;
  const int row  = l & 15;
  const int koff = (l >> 4) << 3;
  const f16* p = s + row * lda + koff;
  v16h a;
#pragma unroll
  for (int h = 0; h < 16; ++h) {
    const int k = ((h >> 3) << 4) + (h & 7);
    a[h] = p[k];
  }
  return a;
}

// B fragment: 32x16 f16 (K x N), k-major source with leading dim ldb.
// lane l: n = l&15; lanes 0-15 hold K = {0..7, 16..23}, lanes 16-31 hold K = {8..15, 24..31}
__device__ __forceinline__ v16h frag_load_B(const f16* __restrict__ s, int ldb) {
  const int l     = threadIdx.x & 31;
  const int n     = l & 15;
  const int kbase = (l >> 4) << 3;
  const f16* p = s + kbase * ldb + n;
  v16h b;
#pragma unroll
  for (int h = 0; h < 16; ++h) b[h] = p[(((h >> 3) << 4) + (h & 7)) * ldb];
  return b;
}

__device__ __forceinline__ v8f wmma_f16(v16h a, v16h b, v8f c) {
  return __builtin_amdgcn_wmma_f32_16x16x32_f16(false, a, false, b, (short)0, c,
                                                false, false);
}

// ---------------------------------------------------------------------------
// K0: Wb (512x512 f32) -> f16
__global__ __launch_bounds__(256) void k_cvt(const float* __restrict__ Wb,
                                             f16* __restrict__ Wb16) {
  int i = blockIdx.x * 256 + threadIdx.x;
  Wb16[i] = (f16)Wb[i];
}

// ---------------------------------------------------------------------------
// K1: tanh RNN over 32 steps. Each wave owns a 16-row tile, N=64, K=64.
// Per step: C = seq*Wih + bias; C += H @ Whh^T (2 K-steps x 4 N-tiles WMMA);
// branch-free tanh; recirculate f16 H through per-wave LDS (wave-private, so
// no workgroup barrier needed: same-wave DS ops are in-order / DScnt-tracked).
__global__ __launch_bounds__(256) void k_rnn(const float* __restrict__ x,
                                             const float* __restrict__ Wih,
                                             const float* __restrict__ Whh,
                                             const float* __restrict__ bih,
                                             const float* __restrict__ bhh,
                                             float* __restrict__ lh32,
                                             f16* __restrict__ lh16) {
  __shared__ f16   sWhhT[64 * 64];        // B source: WhhT[k][n] = Whh[n][k]
  __shared__ float sSeq[8][16 * 32];      // per-wave seq tile [m][t]
  __shared__ f16   sH[8][16 * 64];        // per-wave hidden state (f16)

  const int tid  = threadIdx.x;
  const int wave = tid >> 5;
  const int lane = tid & 31;

  for (int i = tid; i < 64 * 64; i += 256) {
    int k = i >> 6, n = i & 63;
    sWhhT[i] = (f16)Whh[n * 64 + k];
  }

  const int rowBase = (blockIdx.x * 8 + wave) * 16;
  const int b  = rowBase >> 9;
  const int m0 = rowBase & 511;

  // seq[m][t] = x[b][t][m0+m]
  for (int i = lane; i < 16 * 32; i += 32) {
    int t = i >> 4, m = i & 15;
    sSeq[wave][m * 32 + t] = x[((b * 32) + t) * 512 + m0 + m];
  }
  for (int i = lane; i < 16 * 64; i += 32) sH[wave][i] = (f16)0.0f;
  __syncthreads();   // only for sWhhT visibility across waves

  // loop-invariant B fragments of Whh^T
  v16h Bf[2][4];
#pragma unroll
  for (int kk = 0; kk < 2; ++kk)
#pragma unroll
    for (int nt = 0; nt < 4; ++nt)
      Bf[kk][nt] = frag_load_B(&sWhhT[(kk * 32) * 64 + nt * 16], 64);

  const int n0 = lane & 15;
  const int mh = (lane >> 4) << 3;
  float wih_n[4], bsum[4];
#pragma unroll
  for (int nt = 0; nt < 4; ++nt) {
    int n = nt * 16 + n0;
    wih_n[nt] = Wih[n];
    bsum[nt]  = bih[n] + bhh[n];
  }

  v8f acc[4];
  for (int t = 0; t < 32; ++t) {
    v16h a0 = frag_load_A(&sH[wave][0], 64);
    v16h a1 = frag_load_A(&sH[wave][32], 64);
    float sv[8];
#pragma unroll
    for (int r = 0; r < 8; ++r) sv[r] = sSeq[wave][(mh + r) * 32 + t];
#pragma unroll
    for (int nt = 0; nt < 4; ++nt) {
      v8f c;
#pragma unroll
      for (int r = 0; r < 8; ++r) c[r] = sv[r] * wih_n[nt] + bsum[nt];
      c = wmma_f16(a0, Bf[0][nt], c);
      c = wmma_f16(a1, Bf[1][nt], c);
#pragma unroll
      for (int r = 0; r < 8; ++r) c[r] = tanh_fast(c[r]);
      acc[nt] = c;
    }
    __builtin_amdgcn_wave_barrier();   // keep store/load order in scheduler
#pragma unroll
    for (int nt = 0; nt < 4; ++nt)
#pragma unroll
      for (int r = 0; r < 8; ++r)
        sH[wave][(mh + r) * 64 + nt * 16 + n0] = (f16)acc[nt][r];
    __builtin_amdgcn_wave_barrier();
  }

#pragma unroll
  for (int nt = 0; nt < 4; ++nt)
#pragma unroll
    for (int r = 0; r < 8; ++r) {
      int row = rowBase + mh + r;
      int n   = nt * 16 + n0;
      lh32[row * 64 + n] = acc[nt][r];
      lh16[row * 64 + n] = (f16)acc[nt][r];
    }
}

// ---------------------------------------------------------------------------
// K2: e12 = last_hid @ [W1^T | W2^T]  ([8192x64] @ [64x64], WMMA).
// cols 0..31 = e1 (W1), cols 32..63 = e2 (W2)
__global__ __launch_bounds__(256) void k_e12(const f16* __restrict__ lh16,
                                             const float* __restrict__ W1,
                                             const float* __restrict__ W2,
                                             float* __restrict__ e12) {
  __shared__ f16 sB[64 * 64];
  __shared__ f16 sA[8][16 * 64];
  const int tid = threadIdx.x, wave = tid >> 5, lane = tid & 31;

  for (int i = tid; i < 64 * 64; i += 256) {
    int k = i >> 6, n = i & 63;
    sB[i] = (f16)(n < 32 ? W1[n * 64 + k] : W2[(n - 32) * 64 + k]);
  }
  const int rowBase = (blockIdx.x * 8 + wave) * 16;
  const f16* src = lh16 + (size_t)rowBase * 64;
  for (int i = lane; i < 16 * 64; i += 32) sA[wave][i] = src[i];
  __syncthreads();

  v16h a0 = frag_load_A(&sA[wave][0], 64);
  v16h a1 = frag_load_A(&sA[wave][32], 64);
  const int n0 = lane & 15, mh = (lane >> 4) << 3;
#pragma unroll
  for (int nt = 0; nt < 4; ++nt) {
    v8f c = {0.f, 0.f, 0.f, 0.f, 0.f, 0.f, 0.f, 0.f};
    c = wmma_f16(a0, frag_load_B(&sB[nt * 16], 64), c);
    c = wmma_f16(a1, frag_load_B(&sB[32 * 64 + nt * 16], 64), c);
#pragma unroll
    for (int r = 0; r < 8; ++r)
      e12[(size_t)(rowBase + mh + r) * 64 + nt * 16 + n0] = c[r];
  }
}

// ---------------------------------------------------------------------------
// K3: a_raw[b,i,j] = V . elu(e2[b,i,:] + e1[b,j,:] + b1) + bv
// grid (8192, 2): blockIdx.x = b*512+i, blockIdx.y selects 256 j's.
__global__ __launch_bounds__(256) void k_attn(const float* __restrict__ e12,
                                              const float* __restrict__ b1,
                                              const float* __restrict__ Vv,
                                              const float* __restrict__ bv,
                                              float* __restrict__ a_raw) {
  __shared__ float sE1[256 * 36];   // stride 36 floats: 16B aligned rows
  __shared__ float sE2[32], sB1[32], sV[32];
  const int tid = threadIdx.x;
  const int bi  = blockIdx.x;           // b*512 + i
  const int b   = bi >> 9;
  const int j0  = blockIdx.y * 256;

  if (tid < 32) {
    sE2[tid] = e12[(size_t)bi * 64 + 32 + tid];
    sB1[tid] = b1[tid];
    sV[tid]  = Vv[tid];
  }
  for (int idx = tid; idx < 256 * 32; idx += 256) {
    int jl = idx >> 5, h = idx & 31;
    sE1[jl * 36 + h] = e12[(size_t)(b * 512 + j0 + jl) * 64 + h];
  }
  __syncthreads();

  const float* my = &sE1[tid * 36];
  float acc = bv[0];
#pragma unroll
  for (int h = 0; h < 32; ++h) {
    float t  = sE2[h] + my[h] + sB1[h];
    float el = t > 0.f ? t : (__expf(t) - 1.f);   // branchless select
    acc += sV[h] * el;
  }
  a_raw[(size_t)bi * 512 + j0 + tid] = acc;
}

// ---------------------------------------------------------------------------
// K4: rnorm[b,j] = 1 / max(sqrt(sum_i a_raw[b,i,j]^2), 1e-12)
__global__ __launch_bounds__(256) void k_norm(const float* __restrict__ a_raw,
                                              float* __restrict__ rnorm) {
  int idx = blockIdx.x * 256 + threadIdx.x;   // b*512 + j
  int b = idx >> 9, j = idx & 511;
  const float* col = a_raw + (size_t)b * 512 * 512 + j;
  float s = 0.f;
  for (int i = 0; i < 512; ++i) {
    float v = col[(size_t)i * 512];
    s += v * v;
  }
  rnorm[idx] = 1.f / fmaxf(sqrtf(s), 1e-12f);
}

// ---------------------------------------------------------------------------
// K5: fused: cpre = (a_raw * rnorm_col) @ Wb ; c = sigmoid(cpre + wb)
//            adjmix = adj*c + a_norm*(1-c)   -> f16
// block tile 128x64 (8 waves: 4 along M x 2 along N, each 32x32), K=512
__global__ __launch_bounds__(256) void k_gemm_c(const float* __restrict__ a_raw,
                                                const float* __restrict__ rnorm,
                                                const f16* __restrict__ Wb16,
                                                const float* __restrict__ adj,
                                                const float* __restrict__ wb,
                                                f16* __restrict__ adjmix) {
  __shared__ f16 sA[128 * 32];
  __shared__ f16 sB[32 * 64];
  const int tid = threadIdx.x, wave = tid >> 5, lane = tid & 31;
  const int b  = blockIdx.z;
  const int i0 = blockIdx.x * 128;
  const int j0 = blockIdx.y * 64;
  const size_t abase = (size_t)b * 512 * 512;
  const float* Ab = a_raw + abase;
  const float* rn = rnorm + b * 512;
  const int mw = wave >> 1, nw = wave & 1;

  v8f acc[2][2];
#pragma unroll
  for (int p = 0; p < 2; ++p)
#pragma unroll
    for (int q = 0; q < 2; ++q)
      acc[p][q] = (v8f){0.f, 0.f, 0.f, 0.f, 0.f, 0.f, 0.f, 0.f};

  for (int kb = 0; kb < 16; ++kb) {
    const int k0 = kb * 32;
    for (int idx = tid; idx < 128 * 32; idx += 256) {
      int i = idx >> 5, k = idx & 31;
      sA[idx] = (f16)(Ab[(size_t)(i0 + i) * 512 + k0 + k] * rn[k0 + k]);
    }
    for (int idx = tid; idx < 32 * 64; idx += 256) {
      int k = idx >> 6, n = idx & 63;
      sB[idx] = Wb16[(size_t)(k0 + k) * 512 + j0 + n];
    }
    if (kb + 1 < 16)
      __builtin_prefetch(&Ab[(size_t)(i0 + (tid >> 1)) * 512 + k0 + 32], 0, 1);
    __syncthreads();
    v16h af[2], bf[2];
#pragma unroll
    for (int p = 0; p < 2; ++p)
      af[p] = frag_load_A(&sA[(mw * 32 + p * 16) * 32], 32);
#pragma unroll
    for (int q = 0; q < 2; ++q)
      bf[q] = frag_load_B(&sB[nw * 32 + q * 16], 64);
#pragma unroll
    for (int p = 0; p < 2; ++p)
#pragma unroll
      for (int q = 0; q < 2; ++q)
        acc[p][q] = wmma_f16(af[p], bf[q], acc[p][q]);
    __syncthreads();
  }

  const int n0 = lane & 15, mh = (lane >> 4) << 3;
  const float wb0 = wb[0];
#pragma unroll
  for (int p = 0; p < 2; ++p)
#pragma unroll
    for (int q = 0; q < 2; ++q)
#pragma unroll
      for (int r = 0; r < 8; ++r) {
        int i = i0 + mw * 32 + p * 16 + mh + r;
        int j = j0 + nw * 32 + q * 16 + n0;
        float cg  = sigmoid_fast(acc[p][q][r] + wb0);
        float amn = Ab[(size_t)i * 512 + j] * rn[j];
        float mix = adj[(size_t)i * 512 + j] * cg + amn * (1.f - cg);
        adjmix[abase + (size_t)i * 512 + j] = (f16)mix;
      }
}

// ---------------------------------------------------------------------------
// K6: per-node convs + relu + (r @ gc1_w) -> rg f16 [8192 x 64]
__global__ __launch_bounds__(256) void k_rg(const float* __restrict__ x,
                                            const float* __restrict__ conv_w,
                                            const float* __restrict__ conv_b,
                                            const float* __restrict__ convl_w,
                                            const float* __restrict__ convl_b,
                                            const float* __restrict__ gc1_w,
                                            f16* __restrict__ rg) {
  __shared__ float sSeq[8][32];
  __shared__ float sR[8][24];
  __shared__ float sCW[8 * 32], sCLW[8 * 16], sG[24 * 64], sCB[8], sCLB[8];
  const int tid = threadIdx.x;
  for (int i = tid; i < 8 * 32; i += 256) sCW[i] = conv_w[i];
  for (int i = tid; i < 8 * 16; i += 256) sCLW[i] = convl_w[i];
  for (int i = tid; i < 24 * 64; i += 256) sG[i] = gc1_w[i];
  if (tid < 8) { sCB[tid] = conv_b[tid]; sCLB[tid] = convl_b[tid]; }

  const int rowBase = blockIdx.x * 8;
  const int b = rowBase >> 9, m0 = rowBase & 511;
  for (int i = tid; i < 8 * 32; i += 256) {
    int t = i >> 3, r2 = i & 7;
    sSeq[r2][t] = x[((b * 32) + t) * 512 + m0 + r2];
  }
  __syncthreads();

  if (tid < 192) {
    int r2 = tid / 24, q = tid % 24;
    int k = q / 3, c = q % 3;
    float s;
    if (c == 0) {                                 // short conv, kernel == W
      s = sCB[k];
      for (int t = 0; t < 32; ++t) s += sSeq[r2][t] * sCW[k * 32 + t];
    } else {                                      // dilated long conv, d=2
      int tp = c - 1;
      s = sCLB[k];
      for (int ii = 0; ii < 16; ++ii)
        s += sSeq[r2][tp + 2 * ii] * sCLW[k * 16 + ii];
    }
    sR[r2][q] = fmaxf(s, 0.f);                    // relu
  }
  __syncthreads();

  for (int i = tid; i < 8 * 64; i += 256) {
    int r2 = i >> 6, n = i & 63;
    float s = 0.f;
#pragma unroll
    for (int q = 0; q < 24; ++q) s += sR[r2][q] * sG[q * 64 + n];
    rg[(size_t)(rowBase + r2) * 64 + n] = (f16)s;
  }
}

// ---------------------------------------------------------------------------
// K7: h1 = relu(adjmix @ rg + gc1_b), per b: [512x512]@[512x64] -> f16
// block tile 64x64: 8 waves (4 along M, 2 along N), each wave 16x32
__global__ __launch_bounds__(256) void k_gemm_h1(const f16* __restrict__ adjmix,
                                                 const f16* __restrict__ rg,
                                                 const float* __restrict__ gc1_b,
                                                 f16* __restrict__ h1) {
  __shared__ f16 sA[64 * 32];
  __shared__ f16 sB[32 * 64];
  const int tid = threadIdx.x, wave = tid >> 5, lane = tid & 31;
  const int b  = blockIdx.z;
  const int i0 = blockIdx.x * 64;
  const f16* Ab = adjmix + (size_t)b * 512 * 512;
  const f16* Bb = rg + (size_t)b * 512 * 64;
  const int mw = wave >> 1, nw = wave & 1;

  v8f acc[2];
  acc[0] = (v8f){0.f, 0.f, 0.f, 0.f, 0.f, 0.f, 0.f, 0.f};
  acc[1] = acc[0];

  for (int kb = 0; kb < 16; ++kb) {
    const int k0 = kb * 32;
    for (int idx = tid; idx < 64 * 32; idx += 256)
      sA[idx] = Ab[(size_t)(i0 + (idx >> 5)) * 512 + k0 + (idx & 31)];
    for (int idx = tid; idx < 32 * 64; idx += 256)
      sB[idx] = Bb[(size_t)(k0 + (idx >> 6)) * 64 + (idx & 63)];
    __syncthreads();
    v16h af = frag_load_A(&sA[(mw * 16) * 32], 32);
#pragma unroll
    for (int q = 0; q < 2; ++q)
      acc[q] = wmma_f16(af, frag_load_B(&sB[nw * 32 + q * 16], 64), acc[q]);
    __syncthreads();
  }

  const int n0 = lane & 15, mh = (lane >> 4) << 3;
#pragma unroll
  for (int q = 0; q < 2; ++q)
#pragma unroll
    for (int r = 0; r < 8; ++r) {
      int i = i0 + mw * 16 + mh + r;
      int n = nw * 32 + q * 16 + n0;
      float v = fmaxf(acc[q][r] + gc1_b[n], 0.f);
      h1[(size_t)(b * 512 + i) * 64 + n] = (f16)v;
    }
}

// ---------------------------------------------------------------------------
// K8: hs = h1 @ gc2_w, N padded 10 -> 16, f16
__global__ __launch_bounds__(256) void k_hs(const f16* __restrict__ h1,
                                            const float* __restrict__ gc2_w,
                                            f16* __restrict__ hs) {
  int idx = blockIdx.x * 256 + threadIdx.x;   // row*16 + p
  int row = idx >> 4, p = idx & 15;
  float s = 0.f;
  if (p < 10) {
#pragma unroll 8
    for (int k = 0; k < 64; ++k) s += (float)h1[(size_t)row * 64 + k] * gc2_w[k * 10 + p];
  }
  hs[idx] = (f16)s;
}

// ---------------------------------------------------------------------------
// K9: out_spatial = relu(adjmix @ hs + gc2_b), per b: [512x512]@[512x16]
// block tile 128x16: 8 waves, each owns one 16x16 M-subtile
__global__ __launch_bounds__(256) void k_gemm_os(const f16* __restrict__ adjmix,
                                                 const f16* __restrict__ hs,
                                                 const float* __restrict__ gc2_b,
                                                 float* __restrict__ osp) {
  __shared__ f16 sA[128 * 32];
  __shared__ f16 sB[32 * 16];
  const int tid = threadIdx.x, wave = tid >> 5, lane = tid & 31;
  const int b  = blockIdx.z;
  const int i0 = blockIdx.x * 128;
  const f16* Ab = adjmix + (size_t)b * 512 * 512;
  const f16* Bb = hs + (size_t)b * 512 * 16;

  v8f acc = (v8f){0.f, 0.f, 0.f, 0.f, 0.f, 0.f, 0.f, 0.f};
  for (int kb = 0; kb < 16; ++kb) {
    const int k0 = kb * 32;
    for (int idx = tid; idx < 128 * 32; idx += 256)
      sA[idx] = Ab[(size_t)(i0 + (idx >> 5)) * 512 + k0 + (idx & 31)];
    for (int idx = tid; idx < 32 * 16; idx += 256)
      sB[idx] = Bb[(size_t)(k0 + (idx >> 4)) * 16 + (idx & 15)];
    __syncthreads();
    acc = wmma_f16(frag_load_A(&sA[(wave * 16) * 32], 32),
                   frag_load_B(&sB[0], 16), acc);
    __syncthreads();
  }

  const int n0 = lane & 15, mh = (lane >> 4) << 3;
#pragma unroll
  for (int r = 0; r < 8; ++r) {
    int i = i0 + wave * 16 + mh + r;
    float v = (n0 < 10) ? fmaxf(acc[r] + gc2_b[n0], 0.f) : 0.f;
    osp[(size_t)(b * 512 + i) * 16 + n0] = v;
  }
}

// ---------------------------------------------------------------------------
// K10: out[b,h,m] = out_b[h] + [osp(10) | lh(64)] . out_w[h,:]
__global__ __launch_bounds__(256) void k_out(const float* __restrict__ osp,
                                             const float* __restrict__ lh32,
                                             const float* __restrict__ out_w,
                                             const float* __restrict__ out_b,
                                             float* __restrict__ out) {
  __shared__ float sW[8 * 74], sBv[8];
  const int tid = threadIdx.x;
  for (int i = tid; i < 8 * 74; i += 256) sW[i] = out_w[i];
  if (tid < 8) sBv[tid] = out_b[tid];
  __syncthreads();

  const int row = blockIdx.x * 256 + tid;   // b*512 + m
  const int b = row >> 9, m = row & 511;
  float f[74];
#pragma unroll
  for (int p = 0; p < 10; ++p) f[p] = osp[(size_t)row * 16 + p];
#pragma unroll
  for (int n = 0; n < 64; ++n) f[10 + n] = lh32[(size_t)row * 64 + n];
#pragma unroll
  for (int h = 0; h < 8; ++h) {
    float s = sBv[h];
#pragma unroll
    for (int q = 0; q < 74; ++q) s += f[q] * sW[h * 74 + q];
    out[((size_t)(b * 8) + h) * 512 + m] = s;
  }
}

// ---------------------------------------------------------------------------
extern "C" void kernel_launch(void* const* d_in, const int* in_sizes, int n_in,
                              void* d_out, int out_size, void* d_ws, size_t ws_size,
                              hipStream_t stream) {
  (void)in_sizes; (void)n_in; (void)out_size; (void)ws_size;
  const float* x       = (const float*)d_in[0];
  const float* adj     = (const float*)d_in[1];
  const float* Wih     = (const float*)d_in[2];
  const float* Whh     = (const float*)d_in[3];
  const float* bih     = (const float*)d_in[4];
  const float* bhh     = (const float*)d_in[5];
  const float* W1      = (const float*)d_in[6];
  const float* W2      = (const float*)d_in[7];
  const float* b1      = (const float*)d_in[8];
  const float* Vv      = (const float*)d_in[9];
  const float* bv      = (const float*)d_in[10];
  const float* Wb      = (const float*)d_in[11];
  const float* wb      = (const float*)d_in[12];
  const float* conv_w  = (const float*)d_in[13];
  const float* conv_b  = (const float*)d_in[14];
  const float* convl_w = (const float*)d_in[15];
  const float* convl_b = (const float*)d_in[16];
  const float* gc1_w   = (const float*)d_in[17];
  const float* gc1_b   = (const float*)d_in[18];
  const float* gc2_w   = (const float*)d_in[19];
  const float* gc2_b   = (const float*)d_in[20];
  const float* out_w   = (const float*)d_in[21];
  const float* out_b   = (const float*)d_in[22];
  float* outp = (float*)d_out;

  // workspace carve-out (~34 MB)
  size_t off = 0;
  char* base = (char*)d_ws;
  auto take = [&](size_t bytes) -> void* {
    void* p = base + off;
    off += (bytes + 255) & ~(size_t)255;
    return p;
  };
  float* lh32   = (float*)take((size_t)ROWSc * 64 * 4);
  f16*   lh16   = (f16*)  take((size_t)ROWSc * 64 * 2);
  float* e12    = (float*)take((size_t)ROWSc * 64 * 4);
  float* a_raw  = (float*)take((size_t)Bc * 512 * 512 * 4);
  float* rnorm  = (float*)take((size_t)ROWSc * 4);
  f16*   Wb16   = (f16*)  take((size_t)512 * 512 * 2);
  f16*   adjmix = (f16*)  take((size_t)Bc * 512 * 512 * 2);
  f16*   rg     = (f16*)  take((size_t)ROWSc * 64 * 2);
  f16*   h1     = (f16*)  take((size_t)ROWSc * 64 * 2);
  f16*   hs     = (f16*)  take((size_t)ROWSc * 16 * 2);
  float* osp    = (float*)take((size_t)ROWSc * 16 * 4);

  k_cvt    <<<1024, 256, 0, stream>>>(Wb, Wb16);
  k_rnn    <<<64,   256, 0, stream>>>(x, Wih, Whh, bih, bhh, lh32, lh16);
  k_e12    <<<64,   256, 0, stream>>>(lh16, W1, W2, e12);
  k_attn   <<<dim3(8192, 2), 256, 0, stream>>>(e12, b1, Vv, bv, a_raw);
  k_norm   <<<32,   256, 0, stream>>>(a_raw, rnorm);
  k_gemm_c <<<dim3(4, 8, 16), 256, 0, stream>>>(a_raw, rnorm, Wb16, adj, wb, adjmix);
  k_rg     <<<1024, 256, 0, stream>>>(x, conv_w, conv_b, convl_w, convl_b, gc1_w, rg);
  k_gemm_h1<<<dim3(8, 1, 16), 256, 0, stream>>>(adjmix, rg, gc1_b, h1);
  k_hs     <<<512,  256, 0, stream>>>(h1, gc2_w, hs);
  k_gemm_os<<<dim3(4, 1, 16), 256, 0, stream>>>(adjmix, hs, gc2_b, osp);
  k_out    <<<32,   256, 0, stream>>>(osp, lh32, out_w, out_b, outp);
}
